// PixelSNAILBlock_42167988912660
// MI455X (gfx1250) — matrix operand order repaired and checked by reading, not compiled, not run on hardware
//
#include <hip/hip_runtime.h>
#include <cstdint>
#include <cstddef>

// ---------------------------------------------------------------------------
// PixelSNAIL block for MI455X (gfx1250, wave32, WMMA).
// Convs dominate (~103 of ~123 GFLOP) -> bf16 v_wmma_f32_16x16x32_bf16
// implicit GEMM, 64x64 tiles, double-buffered LDS (1 barrier/K-step),
// global_prefetch on the weight stream. Attention is flash-style,
// 32 keys/step, V pre-transposed for wide loads.
// ---------------------------------------------------------------------------

typedef __attribute__((ext_vector_type(16))) __bf16 v16bf;
typedef __attribute__((ext_vector_type(8)))  __bf16 v8bf;
typedef __attribute__((ext_vector_type(8)))  float  v8f;
typedef __attribute__((ext_vector_type(4)))  unsigned int v4u;

union F16 { v16bf bf; v4u u4[2]; };
union F8  { v8bf  bf; v4u u4;    };

#define B_    2
#define CIN_  256
#define P_    4096              // T*H*W = 16*16*16
#define M_    (B_ * P_)         // 8192 GEMM rows (positions)
#define NRES_ 4
#define KC_   16
#define VC_   128
#define CPE_  262

__device__ __forceinline__ float elu_f(float x) {
    return (x > 0.0f) ? x : (__expf(x) - 1.0f);
}

// ---------------------------------------------------------------------------
// small elementwise kernels
// ---------------------------------------------------------------------------
__global__ void cast_kernel(const float* __restrict__ src,
                            __bf16* __restrict__ dst, int n) {
    int i = blockIdx.x * 256 + threadIdx.x;
    if (i < n) dst[i] = (__bf16)src[i];
}

__global__ void elu_cast_kernel(const float* __restrict__ src,
                                __bf16* __restrict__ dst, int n) {
    int i = blockIdx.x * 256 + threadIdx.x;
    if (i < n) dst[i] = (__bf16)elu_f(src[i]);
}

// res [B][256][P] f32 -> [B*P][256] bf16 with ELU (A-matrix for 1x1 GEMM)
__global__ void elu_transpose_kernel(const float* __restrict__ res,
                                     __bf16* __restrict__ dst) {
    int i = blockIdx.x * 256 + threadIdx.x;
    if (i >= M_ * CIN_) return;
    int b = i >> 20;
    int c = (i >> 12) & 255;
    int p = i & 4095;
    dst[((size_t)(b * P_ + p)) * CIN_ + c] = (__bf16)elu_f(res[i]);
}

// res += o[:, :256] * sigmoid(o[:, 256:]);  also emits elu(res) as bf16
// (fused input cast for the next residual block's conv1)
__global__ void gate_add_kernel(float* __restrict__ res,
                                const float* __restrict__ o,
                                __bf16* __restrict__ act) {
    int i = blockIdx.x * 256 + threadIdx.x;
    if (i >= M_ * CIN_) return;
    int b = i >> 20;
    int rem = i & ((1 << 20) - 1);
    float a = o[(size_t)b * 512 * P_ + rem];
    float g = o[(size_t)b * 512 * P_ + 256 * P_ + rem];
    float nv = res[i] + a * (1.0f / (1.0f + __expf(-g)));
    res[i] = nv;
    act[i] = (__bf16)elu_f(nv);
}

__global__ void combine_kernel(const float* __restrict__ p1,
                               const float* __restrict__ p2,
                               __bf16* __restrict__ dst, int n) {
    int i = blockIdx.x * 256 + threadIdx.x;
    if (i < n) dst[i] = (__bf16)elu_f(p1[i] + p2[i]);
}

// ---------------------------------------------------------------------------
// q/k/v projection. q pre-scaled by 1/sqrt(259).
// q,k: [B][pos][16] bf16.  v: [B][col][pos] bf16 (TRANSPOSED for attention).
// ---------------------------------------------------------------------------
__global__ __launch_bounds__(160)
void qkv_kernel(const float* __restrict__ res, const float* __restrict__ orig,
                const float* __restrict__ qw, const float* __restrict__ qb,
                const float* __restrict__ kw, const float* __restrict__ kb,
                const float* __restrict__ vw, const float* __restrict__ vb,
                __bf16* __restrict__ q, __bf16* __restrict__ k,
                __bf16* __restrict__ vt) {
    __shared__ float xa[CPE_];
    int pos = blockIdx.x;
    int b = pos >> 12, p = pos & 4095;
    for (int c = threadIdx.x; c < 256; c += 160)
        xa[c] = res[((size_t)(b * 256 + c)) * P_ + p];
    if (threadIdx.x < 3)
        xa[256 + threadIdx.x] = orig[((size_t)(b * 3 + threadIdx.x)) * P_ + p];
    if (threadIdx.x == 0) {
        int t = p >> 8, h = (p >> 4) & 15, w = p & 15;
        xa[259] = -0.5f + t * 0.0625f;
        xa[260] = -0.5f + h * 0.0625f;
        xa[261] = -0.5f + w * 0.0625f;
    }
    __syncthreads();
    int o = threadIdx.x;
    const float* wrow;
    float bb, scale = 1.0f;
    __bf16* dst;
    if (o < 16) {
        wrow = qw + o * CPE_; bb = qb[o];
        dst = q + (size_t)pos * KC_ + o;
        scale = 0.06213698f;                       // 1/sqrt(259)
    } else if (o < 32) {
        wrow = kw + (o - 16) * CPE_; bb = kb[o - 16];
        dst = k + (size_t)pos * KC_ + (o - 16);
    } else {
        wrow = vw + (o - 32) * CPE_; bb = vb[o - 32];
        dst = vt + ((size_t)(b * VC_ + (o - 32))) * P_ + p;   // transposed
    }
    float s = bb;
    for (int c = 0; c < CPE_; ++c) s += xa[c] * wrow[c];
    *dst = (__bf16)(s * scale);
}

// ---------------------------------------------------------------------------
// Tiled WMMA GEMM: out = act(A @ W^T + bias), W given row-major [N][K] bf16.
// 256 threads = 8 waves, 64x64 tile, K-step 32; each wave: 1 A frag, 2 WMMAs.
// Double-buffered LDS: next tile is fetched to registers during the WMMAs,
// one barrier per K-step. global_prefetch 2 steps ahead on the streams.
// GATHER=1: A is on-the-fly im2col of [B][Cin][16][16][16] (2x2x2 window
// {p-1,p}, zeros below 0), K = Cin*8.
// mode bits: 1 = ELU, 2 = bf16 out (else f32), 4 = channel-major [B][N][P]
// ---------------------------------------------------------------------------
template <int GATHER>
__global__ __launch_bounds__(256)
void gemm_wmma_kernel(const __bf16* __restrict__ A, const __bf16* __restrict__ W,
                      const float* __restrict__ bias, void* __restrict__ outp,
                      int K, int Nout, int mode) {
    __shared__ __attribute__((aligned(16))) __bf16 As[2][64][40];
    __shared__ __attribute__((aligned(16))) __bf16 Bs[2][64][40];
    const int tid  = threadIdx.x;
    const int lane = tid & 31;
    const int wave = tid >> 5;
    const int m0 = blockIdx.x * 64;
    const int n0 = blockIdx.y * 64;
    const int mw = (wave >> 1) * 16;       // 0,16,32,48
    const int nw = (wave & 1) * 32;        // 0,32
    const int arow = tid >> 2;             // 0..63
    const int acol = (tid & 3) * 8;        // 0,8,16,24
    const int mm = lane & 15, hi = lane >> 4;

    int gb, gt, gh, gw;
    {
        int m = m0 + arow;
        gb = m >> 12;
        int p = m & 4095;
        gt = p >> 8; gh = (p >> 4) & 15; gw = p & 15;
    }
    const int cin = K >> 3;
    const __bf16* wrow = W + (size_t)(n0 + arow) * K + acol;
    const __bf16* amrow = A + (size_t)(m0 + arow) * K + acol;

    // fetch one 8-element chunk of the A tile (im2col gather or direct)
    auto fetchA = [&](int k0) -> F8 {
        F8 h;
        if (GATHER) {
            const int ci = (k0 + acol) >> 3;         // 8 taps share a channel
            const size_t chbase = ((size_t)gb * cin + ci) << 12;
#pragma unroll
            for (int j = 0; j < 8; ++j) {
                int tt = gt - 1 + (j >> 2);
                int hh = gh - 1 + ((j >> 1) & 1);
                int ww = gw - 1 + (j & 1);
                bool ok = ((tt | hh | ww) >= 0);
                __bf16 val = A[chbase + ((tt & 15) << 8) + ((hh & 15) << 4) + (ww & 15)];
                h.bf[j] = ok ? val : (__bf16)0.0f;
            }
        } else {
            h.u4 = *(const v4u*)(amrow + k0);
        }
        return h;
    };

    const int ntiles = K >> 5;
    F8 ra = fetchA(0), rb;
    rb.u4 = *(const v4u*)wrow;
    *(v4u*)&As[0][arow][acol] = ra.u4;
    *(v4u*)&Bs[0][arow][acol] = rb.u4;
    __syncthreads();

    v8f acc0 = {}, acc1 = {};
    for (int i = 0; i < ntiles; ++i) {
        const int cur = i & 1, nxt = cur ^ 1;
        const bool more = (i + 1) < ntiles;
        if (more) {                                   // overlap with WMMAs
            ra = fetchA((i + 1) << 5);
            rb.u4 = *(const v4u*)(wrow + ((i + 1) << 5));
        }
        if (i + 2 < ntiles) {                         // gfx1250 global_prefetch
            __builtin_prefetch(wrow + ((i + 2) << 5), 0, 3);
            if (!GATHER) __builtin_prefetch(amrow + ((i + 2) << 5), 0, 3);
        }

        // A 16x32 frag: k = hi*16 + j, contiguous
        F16 af, bf0, bf1;
        af.u4[0] = *(const v4u*)&As[cur][mw + mm][hi * 16];
        af.u4[1] = *(const v4u*)&As[cur][mw + mm][hi * 16 + 8];
        // B 32x16 frag from [n][k] tile: k = hi*16 + j, contiguous
        bf0.u4[0] = *(const v4u*)&Bs[cur][nw + mm][hi * 16];
        bf0.u4[1] = *(const v4u*)&Bs[cur][nw + mm][hi * 16 + 8];
        bf1.u4[0] = *(const v4u*)&Bs[cur][nw + 16 + mm][hi * 16];
        bf1.u4[1] = *(const v4u*)&Bs[cur][nw + 16 + mm][hi * 16 + 8];
        acc0 = __builtin_amdgcn_wmma_f32_16x16x32_bf16(
            false, af.bf, false, bf0.bf, (short)0, acc0, false, false);
        acc1 = __builtin_amdgcn_wmma_f32_16x16x32_bf16(
            false, af.bf, false, bf1.bf, (short)0, acc1, false, false);

        if (more) {
            *(v4u*)&As[nxt][arow][acol] = ra.u4;
            *(v4u*)&Bs[nxt][arow][acol] = rb.u4;
        }
        __syncthreads();                              // single barrier/K-step
    }

    // Epilogue. D layout: VGPR r -> row r + hi*8, col = lane&15.
    const int mbase = m0 + mw + hi * 8;               // 8 consecutive rows
    const int bidx = mbase >> 12, pbase = mbase & 4095;
#pragma unroll
    for (int t = 0; t < 2; ++t) {
        const v8f& acc = t ? acc1 : acc0;
        const int n = n0 + nw + t * 16 + mm;
        const float bv = bias[n];
        float vals[8];
#pragma unroll
        for (int r = 0; r < 8; ++r) {
            float val = acc[r] + bv;
            vals[r] = (mode & 1) ? elu_f(val) : val;
        }
        if (mode & 4) {
            size_t idx = ((size_t)(bidx * Nout + n)) * P_ + pbase;
            if (mode & 2) {
                F8 pk;
#pragma unroll
                for (int r = 0; r < 8; ++r) pk.bf[r] = (__bf16)vals[r];
                *(v4u*)((__bf16*)outp + idx) = pk.u4;
            } else {
                float* op = (float*)outp + idx;
                *(float4*)(op)     = make_float4(vals[0], vals[1], vals[2], vals[3]);
                *(float4*)(op + 4) = make_float4(vals[4], vals[5], vals[6], vals[7]);
            }
        } else {
            float* op = (float*)outp + (size_t)mbase * Nout + n;
#pragma unroll
            for (int r = 0; r < 8; ++r) op[(size_t)r * Nout] = vals[r];
        }
    }
}

// ---------------------------------------------------------------------------
// Flash-style causal attention. One wave per 16-query tile (grid = B*256).
// 32 keys per step: 2 score WMMAs + 8 PV WMMAs (P is 16x32 dense).
// vt is [B][col][pos] so V fragments are 2x b128 loads.
// ---------------------------------------------------------------------------
__global__ __launch_bounds__(32)
void attn_kernel(const __bf16* __restrict__ q, const __bf16* __restrict__ k,
                 const __bf16* __restrict__ vt, float* __restrict__ att) {
    __shared__ __attribute__((aligned(16))) __bf16 pLDS[16][32];
    const int lane = threadIdx.x;
    const int m = lane & 15, hi = lane >> 4;
    const int tile = blockIdx.x;
    const int b = tile >> 8;
    const int q0 = (tile & 255) << 4;
    const v4u z4 = {0u, 0u, 0u, 0u};

    // q fragment (A 16x32): real dims 0..15, upper k zero
    F16 qf;
    qf.u4[0] = hi ? z4 : *(const v4u*)(q + ((size_t)(b * P_ + q0 + m)) * KC_);
    qf.u4[1] = hi ? z4 : *(const v4u*)(q + ((size_t)(b * P_ + q0 + m)) * KC_ + 8);

    float mrow[8], lrow[8];
    v8f acc[8];
    v8f zero8 = {};
#pragma unroll
    for (int r = 0; r < 8; ++r) { mrow[r] = -3.0e38f; lrow[r] = 0.0f; }
#pragma unroll
    for (int c = 0; c < 8; ++c) acc[c] = zero8;

    const int ktmax = q0 >> 5;
    for (int kt = 0; kt <= ktmax; ++kt) {
        const int kbase = kt << 5;
        // two K^T fragments (B 32x16): lanes 0-15 carry dims 0..15
        F16 kf0, kf1;
        if (hi == 0) {
            const __bf16* kr0 = k + ((size_t)(b * P_ + kbase + m)) * KC_;
            const __bf16* kr1 = kr0 + 16 * KC_;
            kf0.u4[0] = *(const v4u*)kr0; kf0.u4[1] = *(const v4u*)(kr0 + 8);
            kf1.u4[0] = *(const v4u*)kr1; kf1.u4[1] = *(const v4u*)(kr1 + 8);
        } else {
            kf0.u4[0] = z4; kf0.u4[1] = z4;
            kf1.u4[0] = z4; kf1.u4[1] = z4;
        }
        v8f s0 = __builtin_amdgcn_wmma_f32_16x16x32_bf16(
            false, qf.bf, false, kf0.bf, (short)0, zero8, false, false);
        v8f s1 = __builtin_amdgcn_wmma_f32_16x16x32_bf16(
            false, qf.bf, false, kf1.bf, (short)0, zero8, false, false);

        const int key0 = kbase + m, key1 = kbase + 16 + m;
#pragma unroll
        for (int r = 0; r < 8; ++r) {
            const int qm = r + hi * 8;
            const int qglob = q0 + qm;
            const bool va = key0 < qglob, vb = key1 < qglob;
            float sv0 = va ? s0[r] : -3.0e38f;
            float sv1 = vb ? s1[r] : -3.0e38f;
            float mx = fmaxf(sv0, sv1);
#pragma unroll
            for (int off = 1; off < 16; off <<= 1)
                mx = fmaxf(mx, __shfl_xor(mx, off, 32));
            const float nm = fmaxf(mrow[r], mx);
            const float p0 = va ? __expf(sv0 - nm) : 0.0f;
            const float p1 = vb ? __expf(sv1 - nm) : 0.0f;
            float sum = p0 + p1;
#pragma unroll
            for (int off = 1; off < 16; off <<= 1)
                sum += __shfl_xor(sum, off, 32);
            const float fac = (mrow[r] >= nm) ? 1.0f : __expf(mrow[r] - nm);
            lrow[r] = lrow[r] * fac + sum;
            mrow[r] = nm;
#pragma unroll
            for (int c = 0; c < 8; ++c) acc[c][r] *= fac;
            pLDS[qm][m] = (__bf16)p0;
            pLDS[qm][m + 16] = (__bf16)p1;
        }
        __syncthreads();
        // P as A 16x32 fragment (all 32 k real now)
        F16 pf;
        pf.u4[0] = *(const v4u*)&pLDS[m][hi * 16];
        pf.u4[1] = *(const v4u*)&pLDS[m][hi * 16 + 8];

#pragma unroll
        for (int c = 0; c < 8; ++c) {
            // V B-fragment: k = j + hi*16 (keys), n = col; vt[col][pos]
            const __bf16* vr = vt + ((size_t)(b * VC_ + c * 16 + m)) * P_ + kbase + hi * 16;
            F16 vf;
            vf.u4[0] = *(const v4u*)vr;
            vf.u4[1] = *(const v4u*)(vr + 8);
            acc[c] = __builtin_amdgcn_wmma_f32_16x16x32_bf16(
                false, pf.bf, false, vf.bf, (short)0, acc[c], false, false);
        }
        __syncthreads();
    }

    // att [B][P][128]; rows with no valid keys -> 0
#pragma unroll
    for (int c = 0; c < 8; ++c) {
#pragma unroll
        for (int r = 0; r < 8; ++r) {
            const int qm = r + hi * 8;
            const float val = (lrow[r] > 0.0f) ? acc[c][r] / lrow[r] : 0.0f;
            att[((size_t)(b * P_ + q0 + qm)) * VC_ + c * 16 + m] = val;
        }
    }
}

// ---------------------------------------------------------------------------
extern "C" void kernel_launch(void* const* d_in, const int* in_sizes, int n_in,
                              void* d_out, int out_size, void* d_ws, size_t ws_size,
                              hipStream_t stream) {
    (void)in_sizes; (void)n_in; (void)out_size; (void)ws_size;
    const float* x     = (const float*)d_in[0];
    const float* orig  = (const float*)d_in[1];
    const float* rbw1  = (const float*)d_in[2];
    const float* rbb1  = (const float*)d_in[3];
    const float* rbw2  = (const float*)d_in[4];
    const float* rbb2  = (const float*)d_in[5];
    const float* qw    = (const float*)d_in[6];
    const float* qb    = (const float*)d_in[7];
    const float* kw    = (const float*)d_in[8];
    const float* kb    = (const float*)d_in[9];
    const float* vw    = (const float*)d_in[10];
    const float* vb    = (const float*)d_in[11];
    const float* attnw = (const float*)d_in[12];
    const float* attnb = (const float*)d_in[13];
    const float* rescw = (const float*)d_in[14];
    const float* rescb = (const float*)d_in[15];
    const float* outw  = (const float*)d_in[16];
    const float* outb  = (const float*)d_in[17];

    char* base = (char*)d_ws;
    size_t off = 0;
    auto carve = [&](size_t bytes) -> void* {
        void* p = base + off;
        off = (off + bytes + 255) & ~(size_t)255;
        return p;
    };
    __bf16* w1c   = (__bf16*)carve((size_t)NRES_ * 256 * 2048 * 2);   // [N][K] bf16
    __bf16* w2c   = (__bf16*)carve((size_t)NRES_ * 512 * 2048 * 2);
    __bf16* awc   = (__bf16*)carve((size_t)256 * 128 * 2);
    __bf16* rwc   = (__bf16*)carve((size_t)256 * 256 * 2);
    __bf16* owc   = (__bf16*)carve((size_t)256 * 256 * 2);
    float*  resF  = (float*)carve((size_t)M_ * CIN_ * 4);
    __bf16* actbf = (__bf16*)carve((size_t)M_ * CIN_ * 2);
    __bf16* c1bf  = (__bf16*)carve((size_t)M_ * CIN_ * 2);
    float*  c2out = (float*)carve((size_t)M_ * 512 * 4);
    __bf16* qbf   = (__bf16*)carve((size_t)M_ * KC_ * 2);
    __bf16* kbf   = (__bf16*)carve((size_t)M_ * KC_ * 2);
    __bf16* vtbf  = (__bf16*)carve((size_t)M_ * VC_ * 2);
    float*  attF  = (float*)carve((size_t)M_ * VC_ * 4);
    __bf16* attbf = (__bf16*)carve((size_t)M_ * VC_ * 2);
    float*  p1    = (float*)carve((size_t)M_ * CIN_ * 4);
    float*  p2    = (float*)carve((size_t)M_ * CIN_ * 4);
    __bf16* resT  = (__bf16*)carve((size_t)M_ * CIN_ * 2);
    __bf16* sumbf = (__bf16*)carve((size_t)M_ * CIN_ * 2);

    const dim3 blk256(256);
    const int nRC = M_ * CIN_;
    const int gRC = (nRC + 255) / 256;

    // ---- weight cast (keep [N][K] layout) ----
    cast_kernel<<<(NRES_ * 256 * 2048 + 255) / 256, blk256, 0, stream>>>(
        rbw1, w1c, NRES_ * 256 * 2048);
    cast_kernel<<<(NRES_ * 512 * 2048 + 255) / 256, blk256, 0, stream>>>(
        rbw2, w2c, NRES_ * 512 * 2048);
    cast_kernel<<<(256 * 128 + 255) / 256, blk256, 0, stream>>>(attnw, awc, 256 * 128);
    cast_kernel<<<(256 * 256 + 255) / 256, blk256, 0, stream>>>(rescw, rwc, 256 * 256);
    cast_kernel<<<(256 * 256 + 255) / 256, blk256, 0, stream>>>(outw, owc, 256 * 256);

    // ---- residual blocks ----
    hipMemcpyAsync(resF, x, (size_t)nRC * sizeof(float), hipMemcpyDeviceToDevice, stream);
    elu_cast_kernel<<<gRC, blk256, 0, stream>>>(resF, actbf, nRC);   // block 0 input
    for (int i = 0; i < NRES_; ++i) {
        gemm_wmma_kernel<1><<<dim3(M_ / 64, 256 / 64), blk256, 0, stream>>>(
            actbf, w1c + (size_t)i * 256 * 2048, rbb1 + i * 256, c1bf,
            2048, 256, /*ELU|bf16|chan-major*/ 7);
        gemm_wmma_kernel<1><<<dim3(M_ / 64, 512 / 64), blk256, 0, stream>>>(
            c1bf, w2c + (size_t)i * 512 * 2048, rbb2 + i * 512, c2out,
            2048, 512, /*f32 chan-major*/ 4);
        gate_add_kernel<<<gRC, blk256, 0, stream>>>(resF, c2out, actbf); // fused cast
    }

    // ---- attention ----
    qkv_kernel<<<M_, 160, 0, stream>>>(resF, orig, qw, qb, kw, kb, vw, vb,
                                       qbf, kbf, vtbf);
    attn_kernel<<<B_ * 256, 32, 0, stream>>>(qbf, kbf, vtbf, attF);

    // ---- output head: three 1x1 convs ----
    elu_cast_kernel<<<(M_ * VC_ + 255) / 256, blk256, 0, stream>>>(attF, attbf, M_ * VC_);
    gemm_wmma_kernel<0><<<dim3(M_ / 64, 256 / 64), blk256, 0, stream>>>(
        attbf, awc, attnb, p1, 128, 256, /*ELU f32 pos-major*/ 1);
    elu_transpose_kernel<<<gRC, blk256, 0, stream>>>(resF, resT);
    gemm_wmma_kernel<0><<<dim3(M_ / 64, 256 / 64), blk256, 0, stream>>>(
        resT, rwc, rescb, p2, 256, 256, 1);
    combine_kernel<<<gRC, blk256, 0, stream>>>(p1, p2, sumbf, nRC);
    gemm_wmma_kernel<0><<<dim3(M_ / 64, 256 / 64), blk256, 0, stream>>>(
        sumbf, owc, outb, d_out, 256, 256, /*ELU f32 chan-major*/ 5);
}
